// DSCAMSFF_326417515013
// MI455X (gfx1250) — compile-verified, matrix operand order checked by reading, not run on hardware
//
#include <hip/hip_runtime.h>
#include <hip/hip_bf16.h>
#include <math.h>

// ---------------------------------------------------------------------------
// Problem constants (from reference):
//   B=8, CI=2048, CO=512, HW=16*16=256, G=8, CG=256, HID=64
// Only x4/cov4/gce_* are live; cov1-3 paths are dead code in the reference.
// ---------------------------------------------------------------------------

typedef __attribute__((ext_vector_type(16))) __bf16 bf16x16;
typedef __attribute__((ext_vector_type(8)))  __bf16 bf16x8;
typedef __attribute__((ext_vector_type(8)))  float  v8f;
typedef __attribute__((ext_vector_type(4)))  float  v4f;

union BF16Frag { bf16x16 v; bf16x8 h[2]; };

#define TILE_M 64
#define TILE_N 64
#define TILE_K 64
#define LDSK   72   // bf16 elements per row: 144 B (multiple of 16) + pad

__device__ __forceinline__ float sigmoidf_(float x) {
    return 1.0f / (1.0f + __expf(-x));
}

// ---------------------------------------------------------------------------
// Kernel 1: y4[b, co, hw] = sum_ci W[co, ci] * X[b, ci, hw] + bias[co]
// bf16x3 split-precision GEMM on v_wmma_f32_16x16x32_bf16.
// grid: (HW/64, CO/64, B), block: 128 (4 waves); wave -> 32x32 patch.
// ---------------------------------------------------------------------------
__global__ __launch_bounds__(128)
void conv4_gemm_kernel(const float* __restrict__ X,    // [B, 2048, 256]
                       const float* __restrict__ W,    // [512, 2048]
                       const float* __restrict__ bias, // [512]
                       float* __restrict__ Y)          // [B, 512, 256]
{
    const int CI = 2048, CO = 512, HW = 256;
    const int b   = blockIdx.z;
    const int n0  = blockIdx.x * TILE_N;
    const int m0  = blockIdx.y * TILE_M;
    const int tid = threadIdx.x;
    const int lane = tid & 31;
    const int wave = tid >> 5;   // 0..3
    const int wm   = wave >> 1;  // 0..1
    const int wn   = wave & 1;   // 0..1
    const int half = lane >> 4;  // K-group select per ISA 16-bit A/B layout
    const int lr   = lane & 15;

    __shared__ __align__(16) __bf16 Ah[TILE_M][LDSK];
    __shared__ __align__(16) __bf16 Al[TILE_M][LDSK];
    __shared__ __align__(16) __bf16 Bh[TILE_N][LDSK];
    __shared__ __align__(16) __bf16 Bl[TILE_N][LDSK];

    v8f acc[2][2] = {};

    const float* Xb = X + (size_t)b * CI * HW;

    for (int k0 = 0; k0 < CI; k0 += TILE_K) {
        // ---- stage A tile: W[m0+m][k0+kk], split into hi/lo bf16 ----
        {
            const int kk4 = (tid & 15) * 4;
            const int mr  = tid >> 4;
#pragma unroll
            for (int i = 0; i < 8; ++i) {
                const int m = mr + i * 8;
                v4f v = *(const v4f*)(W + (size_t)(m0 + m) * CI + k0 + kk4);
#pragma unroll
                for (int j = 0; j < 4; ++j) {
                    const float f = v[j];
                    const __bf16 hi = (__bf16)f;
                    Ah[m][kk4 + j] = hi;
                    Al[m][kk4 + j] = (__bf16)(f - (float)hi);
                }
            }
        }
        // ---- stage B tile transposed: X[k0+kk][n0+nn] -> Bt[nn][kk] ----
        {
            const int nn4 = (tid & 15) * 4;
            const int kr  = tid >> 4;
#pragma unroll
            for (int i = 0; i < 8; ++i) {
                const int kk = kr + i * 8;
                v4f v = *(const v4f*)(Xb + (size_t)(k0 + kk) * HW + n0 + nn4);
#pragma unroll
                for (int j = 0; j < 4; ++j) {
                    const float f = v[j];
                    const __bf16 hi = (__bf16)f;
                    Bh[nn4 + j][kk] = hi;
                    Bl[nn4 + j][kk] = (__bf16)(f - (float)hi);
                }
            }
        }
        // prefetch next K-tile into caches while we compute this one
        if (k0 + TILE_K < CI) {
            __builtin_prefetch(W + (size_t)(m0 + (tid >> 4)) * CI + k0 + TILE_K + (tid & 15) * 4, 0, 1);
            __builtin_prefetch(Xb + (size_t)(k0 + TILE_K + (tid >> 4)) * HW + n0 + (tid & 15) * 4, 0, 1);
        }
        __syncthreads();

#pragma unroll
        for (int ks = 0; ks < TILE_K; ks += 32) {
            BF16Frag ah[2], al[2], bh[2], bl[2];
#pragma unroll
            for (int i = 0; i < 2; ++i) {
                const int m = wm * 32 + i * 16 + lr;
                ah[i].h[0] = *(const bf16x8*)&Ah[m][ks + 8 * half];
                ah[i].h[1] = *(const bf16x8*)&Ah[m][ks + 16 + 8 * half];
                al[i].h[0] = *(const bf16x8*)&Al[m][ks + 8 * half];
                al[i].h[1] = *(const bf16x8*)&Al[m][ks + 16 + 8 * half];
            }
#pragma unroll
            for (int j = 0; j < 2; ++j) {
                const int n = wn * 32 + j * 16 + lr;
                bh[j].h[0] = *(const bf16x8*)&Bh[n][ks + 8 * half];
                bh[j].h[1] = *(const bf16x8*)&Bh[n][ks + 16 + 8 * half];
                bl[j].h[0] = *(const bf16x8*)&Bl[n][ks + 8 * half];
                bl[j].h[1] = *(const bf16x8*)&Bl[n][ks + 16 + 8 * half];
            }
#pragma unroll
            for (int i = 0; i < 2; ++i)
#pragma unroll
                for (int j = 0; j < 2; ++j) {
                    // fp32 ~= hi*hi + hi*lo + lo*hi  (bf16x3 decomposition)
                    acc[i][j] = __builtin_amdgcn_wmma_f32_16x16x32_bf16(
                        false, ah[i].v, false, bh[j].v, (short)0, acc[i][j], false, false);
                    acc[i][j] = __builtin_amdgcn_wmma_f32_16x16x32_bf16(
                        false, ah[i].v, false, bl[j].v, (short)0, acc[i][j], false, false);
                    acc[i][j] = __builtin_amdgcn_wmma_f32_16x16x32_bf16(
                        false, al[i].v, false, bh[j].v, (short)0, acc[i][j], false, false);
                }
        }
        __syncthreads();
    }

    // ---- store D (+bias): lane L -> N=L%16; VGPR v -> M=v+8*(L/16) ----
    float* Yb = Y + (size_t)b * CO * HW;
#pragma unroll
    for (int i = 0; i < 2; ++i)
#pragma unroll
        for (int j = 0; j < 2; ++j) {
            const int n = n0 + wn * 32 + j * 16 + lr;
#pragma unroll
            for (int v = 0; v < 8; ++v) {
                const int m = m0 + wm * 32 + i * 16 + v + 8 * half;
                Yb[(size_t)m * HW + n] = acc[i][j][v] + bias[m];
            }
        }
}

// ---------------------------------------------------------------------------
// Kernel 2: per-channel spatial mean: avg[b*512+ch] = mean_hw y4[b,ch,hw]
// one block (256 threads) per (b, ch)
// ---------------------------------------------------------------------------
__global__ __launch_bounds__(256)
void chan_mean_kernel(const float* __restrict__ Y, float* __restrict__ avg)
{
    __shared__ float red[256];
    const int row = blockIdx.x;  // b*512 + ch
    const int t = threadIdx.x;
    red[t] = Y[(size_t)row * 256 + t];
    __syncthreads();
    for (int s = 128; s > 0; s >>= 1) {
        if (t < s) red[t] += red[t + s];
        __syncthreads();
    }
    if (t == 0) avg[row] = red[0] * (1.0f / 256.0f);
}

// ---------------------------------------------------------------------------
// Kernel 3: per-(b,g) channel attention:
//   h  = relu(fc1_w[g] @ avg_g + fc1_b[g])      (64 x 256)
//   ca = sigmoid(fc2_w[g] @ h + fc2_b[g])       (256 x 64)
// avg_g[c] = avg[b, 256*(g&1) + c]  (x = concat of y4 x4 along C)
// ---------------------------------------------------------------------------
__global__ __launch_bounds__(256)
void gce_fc_kernel(const float* __restrict__ avg,   // [B,512]
                   const float* __restrict__ fc1_w, // [8,64,256]
                   const float* __restrict__ fc1_b, // [8,64]
                   const float* __restrict__ fc2_w, // [8,256,64]
                   const float* __restrict__ fc2_b, // [8,256]
                   float* __restrict__ ca)          // [B,8,256]
{
    const int bg = blockIdx.x;      // b*8 + g
    const int b = bg >> 3, g = bg & 7;
    const int t = threadIdx.x;
    __shared__ float avL[256];
    __shared__ float hL[64];
    avL[t] = avg[b * 512 + (g & 1) * 256 + t];
    __syncthreads();
    if (t < 64) {
        const float* w = fc1_w + ((size_t)g * 64 + t) * 256;
        float s = fc1_b[g * 64 + t];
        for (int c = 0; c < 256; ++c) s += w[c] * avL[c];
        hL[t] = fmaxf(s, 0.0f);
    }
    __syncthreads();
    {
        const float* w = fc2_w + ((size_t)g * 256 + t) * 64;
        float s = fc2_b[g * 256 + t];
        for (int o = 0; o < 64; ++o) s += w[o] * hL[o];
        ca[(size_t)bg * 256 + t] = sigmoidf_(s);
    }
}

// ---------------------------------------------------------------------------
// Kernel 4: spatial attention + per-group sigmoid sum. Block per (b,g),
// thread t <-> hw. Owns its group entirely -> direct gsum write, no atomics.
// ---------------------------------------------------------------------------
__global__ __launch_bounds__(256)
void gce_sa_kernel(const float* __restrict__ Y,     // [B,512,256]
                   const float* __restrict__ ca,    // [B,8,256]
                   const float* __restrict__ sa_w,  // [8,256]
                   const float* __restrict__ sa_b,  // [8]
                   float* __restrict__ sa,          // [B,8,256]
                   float* __restrict__ gsum)        // [B,8]
{
    const int bg = blockIdx.x;
    const int b = bg >> 3, g = bg & 7;
    const int hw = threadIdx.x;
    __shared__ float caL[256];
    __shared__ float cwL[256];
    __shared__ float red[256];
    const float c0 = ca[(size_t)bg * 256 + hw];
    caL[hw] = c0;
    cwL[hw] = c0 * sa_w[g * 256 + hw];
    __syncthreads();

    const float* Yg = Y + ((size_t)b * 512 + (g & 1) * 256) * 256;

    float acc = 0.0f;
    for (int c = 0; c < 256; ++c)
        acc += Yg[(size_t)c * 256 + hw] * cwL[c];
    const float s = sigmoidf_(acc + sa_b[g]);
    sa[(size_t)bg * 256 + hw] = s;

    float ssum = 0.0f;
    for (int c = 0; c < 256; ++c) {
        const float yv = Yg[(size_t)c * 256 + hw] * caL[c] * s;
        ssum += sigmoidf_(yv);
    }
    red[hw] = ssum;
    __syncthreads();
    for (int st = 128; st > 0; st >>= 1) {
        if (hw < st) red[hw] += red[hw + st];
        __syncthreads();
    }
    if (hw == 0) gsum[bg] = red[0];
}

// ---------------------------------------------------------------------------
// Kernel 5: out[b,ch,hw] = v * (mask + 1), v = y4[b, ch%512, hw]
//   mask = (sig(v*ca*sa) > groupmean) ? 1 : sig(v*ca*sa)
// ---------------------------------------------------------------------------
__global__ __launch_bounds__(256)
void final_gate_kernel(const float* __restrict__ Y,
                       const float* __restrict__ ca,
                       const float* __restrict__ sa,
                       const float* __restrict__ gsum,
                       float* __restrict__ out)
{
    const size_t idx = (size_t)blockIdx.x * blockDim.x + threadIdx.x;
    const int hw = (int)(idx & 255);
    const int ch = (int)((idx >> 8) & 2047);
    const int b  = (int)(idx >> 19);           // 2048*256 = 2^19
    const int g  = ch >> 8;
    const int c  = ch & 255;
    const int ch512 = ch & 511;

    const float v  = Y[((size_t)b * 512 + ch512) * 256 + hw];
    const float m  = gsum[b * 8 + g] * (1.0f / 65536.0f);
    const float yv = v * ca[((size_t)b * 8 + g) * 256 + c]
                       * sa[((size_t)b * 8 + g) * 256 + hw];
    const float s  = sigmoidf_(yv);
    const float mask = (s > m) ? 1.0f : s;
    out[idx] = v * (mask + 1.0f);
}

// ---------------------------------------------------------------------------
extern "C" void kernel_launch(void* const* d_in, const int* in_sizes, int n_in,
                              void* d_out, int out_size, void* d_ws, size_t ws_size,
                              hipStream_t stream) {
    (void)in_sizes; (void)n_in; (void)out_size; (void)ws_size;
    // setup_inputs order: x1,x2,x3,x4, cov1_w,cov1_b, cov2_w,cov2_b,
    // cov3_w,cov3_b, cov4_w,cov4_b, gce_fc1_w,gce_fc1_b, gce_fc2_w,gce_fc2_b,
    // gce_sa_w,gce_sa_b
    const float* x4   = (const float*)d_in[3];
    const float* w4   = (const float*)d_in[10];
    const float* b4   = (const float*)d_in[11];
    const float* fc1w = (const float*)d_in[12];
    const float* fc1b = (const float*)d_in[13];
    const float* fc2w = (const float*)d_in[14];
    const float* fc2b = (const float*)d_in[15];
    const float* saw  = (const float*)d_in[16];
    const float* sab  = (const float*)d_in[17];
    float* out = (float*)d_out;

    // workspace layout (floats)
    float* ws   = (float*)d_ws;
    float* y4   = ws;                 // 8*512*256   = 1,048,576
    float* avg  = y4 + 1048576;       // 8*512       = 4,096
    float* ca   = avg + 4096;         // 8*8*256     = 16,384
    float* sa   = ca + 16384;         // 8*8*256     = 16,384
    float* gsum = sa + 16384;         // 8*8         = 64

    dim3 g1(256 / TILE_N, 512 / TILE_M, 8);   // (4, 8, 8)
    conv4_gemm_kernel<<<g1, 128, 0, stream>>>(x4, w4, b4, y4);
    chan_mean_kernel<<<8 * 512, 256, 0, stream>>>(y4, avg);
    gce_fc_kernel<<<64, 256, 0, stream>>>(avg, fc1w, fc1b, fc2w, fc2b, ca);
    gce_sa_kernel<<<64, 256, 0, stream>>>(y4, ca, saw, sab, sa, gsum);
    final_gate_kernel<<<(8 * 2048 * 256) / 256, 256, 0, stream>>>(y4, ca, sa, gsum, out);
}
